// MultiHeadAttention_82085414961912
// MI455X (gfx1250) — matrix operand order repaired and checked by reading, not compiled, not run on hardware
//
#include <hip/hip_runtime.h>

// ---------- types ----------
typedef __attribute__((ext_vector_type(16))) __bf16 v16bf;
typedef __attribute__((ext_vector_type(8)))  float  v8f;
typedef __attribute__((ext_vector_type(8)))  unsigned short ush8;

#define B_   4
#define N_   2048
#define E_   512
#define H_   8
#define DH_  64
#define M_   (B_ * N_)   // 8192 rows

// ---------- helpers ----------
__device__ __forceinline__ unsigned short f2bf(float f) {
  // round-to-nearest-even fp32 -> bf16
  unsigned u = __float_as_uint(f);
  u += 0x7FFFu + ((u >> 16) & 1u);
  return (unsigned short)(u >> 16);
}

union AB16 { v16bf v; ush8 h[2]; };

// A fragment: 16x32 bf16, row-major source with leading dim ld.
// lane holds row M=lane&15; K = {hi*8 .. hi*8+7} U {16+hi*8 .. 16+hi*8+7}
__device__ __forceinline__ v16bf load_a16(const unsigned short* base, int ld, int lane) {
  const int m = lane & 15, hi = lane >> 4;
  const unsigned short* p = base + (size_t)m * ld;
  AB16 u;
  u.h[0] = *reinterpret_cast<const ush8*>(p + hi * 8);
  u.h[1] = *reinterpret_cast<const ush8*>(p + 16 + hi * 8);
  return u.v;
}

// B fragment where B[k,n] = Src[n,k] (Src row-major, ld = leading dim).
// lane holds col n=lane&15; K = {hi*8 .. hi*8+7} U {16+hi*8 .. 16+hi*8+7} (in Src row n)
__device__ __forceinline__ v16bf load_bt16(const unsigned short* base, int ld, int lane) {
  const int n = lane & 15, hi = lane >> 4;
  const unsigned short* p = base + (size_t)n * ld;
  AB16 u;
  u.h[0] = *reinterpret_cast<const ush8*>(p + hi * 8);
  u.h[1] = *reinterpret_cast<const ush8*>(p + 16 + hi * 8);
  return u.v;
}

__device__ __forceinline__ v8f wmma_bf(v16bf a, v16bf b, v8f c) {
  return __builtin_amdgcn_wmma_f32_16x16x32_bf16(false, a, false, b, (short)0, c, false, false);
}

__device__ __forceinline__ float rmax16(float v) {
  v = fmaxf(v, __shfl_xor(v, 1));
  v = fmaxf(v, __shfl_xor(v, 2));
  v = fmaxf(v, __shfl_xor(v, 4));
  v = fmaxf(v, __shfl_xor(v, 8));
  return v;
}
__device__ __forceinline__ float rsum16(float v) {
  v += __shfl_xor(v, 1);
  v += __shfl_xor(v, 2);
  v += __shfl_xor(v, 4);
  v += __shfl_xor(v, 8);
  return v;
}

// ---------- kernel 1: fp32 -> bf16 convert ----------
__global__ void cvt_bf16_kernel(const float* __restrict__ src,
                                unsigned short* __restrict__ dst, int n) {
  int i = blockIdx.x * blockDim.x + threadIdx.x;
  int stride = gridDim.x * blockDim.x;
  for (; i < n; i += stride) dst[i] = f2bf(src[i]);
}

// ---------- kernel 2: QKV projection  out = x @ W^T + b ----------
// grid (M_/32, 2, 3) block 128; each wave computes a 32x64 tile:
// two A row-fragments share the same four B fragments -> 1.5 b128 loads
// per WMMA instead of 2.5, doubling WMMA density in the schedule.
__global__ __launch_bounds__(128) void proj_qkv_kernel(
    const unsigned short* __restrict__ X,
    const unsigned short* __restrict__ Wq, const unsigned short* __restrict__ Wk,
    const unsigned short* __restrict__ Wv,
    const float* __restrict__ bq, const float* __restrict__ bk,
    const float* __restrict__ bv,
    unsigned short* __restrict__ Qo, unsigned short* __restrict__ Ko,
    unsigned short* __restrict__ Vo) {
  const int lane = threadIdx.x & 31;
  const int w    = threadIdx.x >> 5;
  const int mode = blockIdx.z;
  const int m0    = blockIdx.x * 32;
  const int cbase = (blockIdx.y * 4 + w) * 64;

  const unsigned short* W = (mode == 0) ? Wq : ((mode == 1) ? Wk : Wv);
  const float* bias       = (mode == 0) ? bq : ((mode == 1) ? bk : bv);
  unsigned short* dst     = (mode == 0) ? Qo : ((mode == 1) ? Ko : Vo);
  const float scale = (mode == 0) ? 0.125f : 1.0f;  // 1/sqrt(64) folded into Q

  v8f acc0[4], acc1[4];
  const v8f zf = {0.f, 0.f, 0.f, 0.f, 0.f, 0.f, 0.f, 0.f};
#pragma unroll
  for (int t = 0; t < 4; t++) { acc0[t] = zf; acc1[t] = zf; }

  const unsigned short* Xrow0 = X + (size_t)m0 * E_;
  const unsigned short* Xrow1 = X + (size_t)(m0 + 16) * E_;

#pragma unroll
  for (int kk = 0; kk < E_; kk += 32) {
    v16bf a0 = load_a16(Xrow0 + kk, E_, lane);
    v16bf a1 = load_a16(Xrow1 + kk, E_, lane);
#pragma unroll
    for (int t = 0; t < 4; t++) {
      v16bf bt = load_bt16(W + (size_t)(cbase + t * 16) * E_ + kk, E_, lane);
      acc0[t] = wmma_bf(a0, bt, acc0[t]);
      acc1[t] = wmma_bf(a1, bt, acc1[t]);
    }
  }

  const int hi = lane >> 4, n = lane & 15;
#pragma unroll
  for (int g = 0; g < 2; g++) {
#pragma unroll
    for (int t = 0; t < 4; t++) {
      const int c  = cbase + t * 16 + n;
      const int hh = c >> 6, dh = c & 63;
      const float bc = bias[c];
      const v8f* acc = (g == 0) ? acc0 : acc1;
#pragma unroll
      for (int r = 0; r < 8; r++) {
        const int row = m0 + g * 16 + r + 8 * hi;
        const int bb = row >> 11, nn = row & (N_ - 1);
        const float val = (acc[t][r] + bc) * scale;
        size_t idx;
        if (mode == 2) idx = (((size_t)bb * H_ + hh) * DH_ + dh) * N_ + nn;   // V^T
        else           idx = (((size_t)bb * H_ + hh) * N_ + nn) * DH_ + dh;   // Q,K
        dst[idx] = f2bf(val);
      }
    }
  }
}

// ---------- kernel 3: flash attention ----------
// grid (N_/64, H_, B_) block 128; each wave owns one 16-query tile.
// K fragments are pipelined one key-chunk ahead; V loads are issued right
// after the S-WMMAs so they overlap the softmax VALU work.
__global__ __launch_bounds__(128) void flash_attn_kernel(
    const unsigned short* __restrict__ Qb, const unsigned short* __restrict__ Kb,
    const unsigned short* __restrict__ VTb, unsigned short* __restrict__ attn) {
  const int lane = threadIdx.x & 31;
  const int w    = threadIdx.x >> 5;
  const int b = blockIdx.z, h = blockIdx.y;
  const int q0 = (blockIdx.x * 4 + w) * 16;
  const int bh = b * H_ + h;

  const unsigned short* Qh  = Qb  + (size_t)bh * N_ * DH_;
  const unsigned short* Kh  = Kb  + (size_t)bh * N_ * DH_;
  const unsigned short* VTh = VTb + (size_t)bh * DH_ * N_;

  __shared__ __align__(16) unsigned short pstage[4][16 * 32];
  unsigned short* pl = pstage[w];

  // Q fragments (already scaled by 1/sqrt(Dh))
  const v16bf aq0 = load_a16(Qh + (size_t)q0 * DH_, DH_, lane);
  const v16bf aq1 = load_a16(Qh + (size_t)q0 * DH_ + 32, DH_, lane);

  const v8f zf = {0.f, 0.f, 0.f, 0.f, 0.f, 0.f, 0.f, 0.f};
  v8f o[4];
#pragma unroll
  for (int t = 0; t < 4; t++) o[t] = zf;
  float m_i[8], l_i[8];
#pragma unroll
  for (int r = 0; r < 8; r++) { m_i[r] = -1e30f; l_i[r] = 0.f; }

  const int hi = lane >> 4, n = lane & 15;

  // prologue: K fragments for kb = 0
  // kf[0]: keys 0..15 / dh 0..31, kf[1]: keys 0..15 / dh 32..63
  // kf[2]: keys 16..31 / dh 0..31, kf[3]: keys 16..31 / dh 32..63
  v16bf kf[4];
  kf[0] = load_bt16(Kh, DH_, lane);
  kf[1] = load_bt16(Kh + 32, DH_, lane);
  kf[2] = load_bt16(Kh + 16 * DH_, DH_, lane);
  kf[3] = load_bt16(Kh + 16 * DH_ + 32, DH_, lane);

  for (int kb = 0; kb < N_; kb += 32) {
    // S = Q @ K^T over Dh=64 (two K=32 chunks), two 16-key subtiles
    v8f s0 = zf, s1 = zf;
    s0 = wmma_bf(aq0, kf[0], s0);
    s0 = wmma_bf(aq1, kf[1], s0);
    s1 = wmma_bf(aq0, kf[2], s1);
    s1 = wmma_bf(aq1, kf[3], s1);

    // issue V loads now -> they overlap the softmax VALU work below
    v16bf vf[4];
#pragma unroll
    for (int t = 0; t < 4; t++)
      vf[t] = load_bt16(VTh + (size_t)(t * 16) * N_ + kb, N_, lane);

    // issue next-chunk K loads (address clamped on last iter; result unused)
    const int kn = (kb + 32) & (N_ - 1);
    kf[0] = load_bt16(Kh + (size_t)kn * DH_, DH_, lane);
    kf[1] = load_bt16(Kh + (size_t)kn * DH_ + 32, DH_, lane);
    kf[2] = load_bt16(Kh + (size_t)(kn + 16) * DH_, DH_, lane);
    kf[3] = load_bt16(Kh + (size_t)(kn + 16) * DH_ + 32, DH_, lane);
    __builtin_prefetch(Kh + (size_t)((kb + 64) & (N_ - 1)) * DH_, 0, 0);

    // online softmax update (rows = r + 8*hi, 16 lanes per row)
    float p0[8], p1[8], corr[8];
#pragma unroll
    for (int r = 0; r < 8; r++) {
      float rm = rmax16(fmaxf(s0[r], s1[r]));
      float mnew = fmaxf(m_i[r], rm);
      corr[r] = __expf(m_i[r] - mnew);
      p0[r] = __expf(s0[r] - mnew);
      p1[r] = __expf(s1[r] - mnew);
      float sum = rsum16(p0[r] + p1[r]);
      l_i[r] = l_i[r] * corr[r] + sum;
      m_i[r] = mnew;
    }
#pragma unroll
    for (int t = 0; t < 4; t++)
#pragma unroll
      for (int r = 0; r < 8; r++) o[t][r] *= corr[r];

    // P (C-layout) -> LDS -> A-layout bf16 fragment
#pragma unroll
    for (int r = 0; r < 8; r++) {
      pl[(r + 8 * hi) * 32 + n]      = f2bf(p0[r]);
      pl[(r + 8 * hi) * 32 + 16 + n] = f2bf(p1[r]);
    }
    asm volatile("s_wait_dscnt 0" ::: "memory");
    v16bf pa = load_a16(pl, 32, lane);

    // O += P @ V  (V stored transposed: VT[dh][key])
#pragma unroll
    for (int t = 0; t < 4; t++) o[t] = wmma_bf(pa, vf[t], o[t]);
  }

  // finalize: divide by l, store bf16 into attn [B, N, E]
  float inv[8];
#pragma unroll
  for (int r = 0; r < 8; r++) inv[r] = 1.0f / l_i[r];
  unsigned short* arow = attn + ((size_t)b * N_ + q0) * E_ + h * DH_;
#pragma unroll
  for (int t = 0; t < 4; t++)
#pragma unroll
    for (int r = 0; r < 8; r++)
      arow[(size_t)(r + 8 * hi) * E_ + t * 16 + n] = f2bf(o[t][r] * inv[r]);
}

// ---------- kernel 4: output projection  out = attn @ Wo^T + bo (fp32) ----------
// Same 32x64-tile-per-wave structure as the QKV projection.
__global__ __launch_bounds__(128) void out_proj_kernel(
    const unsigned short* __restrict__ A, const unsigned short* __restrict__ Wo,
    const float* __restrict__ bo, float* __restrict__ out) {
  const int lane = threadIdx.x & 31;
  const int w    = threadIdx.x >> 5;
  const int m0    = blockIdx.x * 32;
  const int cbase = (blockIdx.y * 4 + w) * 64;

  v8f acc0[4], acc1[4];
  const v8f zf = {0.f, 0.f, 0.f, 0.f, 0.f, 0.f, 0.f, 0.f};
#pragma unroll
  for (int t = 0; t < 4; t++) { acc0[t] = zf; acc1[t] = zf; }

  const unsigned short* Arow0 = A + (size_t)m0 * E_;
  const unsigned short* Arow1 = A + (size_t)(m0 + 16) * E_;

#pragma unroll
  for (int kk = 0; kk < E_; kk += 32) {
    v16bf a0 = load_a16(Arow0 + kk, E_, lane);
    v16bf a1 = load_a16(Arow1 + kk, E_, lane);
#pragma unroll
    for (int t = 0; t < 4; t++) {
      v16bf bt = load_bt16(Wo + (size_t)(cbase + t * 16) * E_ + kk, E_, lane);
      acc0[t] = wmma_bf(a0, bt, acc0[t]);
      acc1[t] = wmma_bf(a1, bt, acc1[t]);
    }
  }

  const int hi = lane >> 4, n = lane & 15;
#pragma unroll
  for (int g = 0; g < 2; g++) {
    const v8f* acc = (g == 0) ? acc0 : acc1;
#pragma unroll
    for (int t = 0; t < 4; t++) {
      const int c = cbase + t * 16 + n;
      const float bc = bo[c];
#pragma unroll
      for (int r = 0; r < 8; r++) {
        const int row = m0 + g * 16 + r + 8 * hi;
        out[(size_t)row * E_ + c] = acc[t][r] + bc;
      }
    }
  }
}

// ---------- launch ----------
extern "C" void kernel_launch(void* const* d_in, const int* in_sizes, int n_in,
                              void* d_out, int out_size, void* d_ws, size_t ws_size,
                              hipStream_t stream) {
  const float* q  = (const float*)d_in[0];
  const float* wq = (const float*)d_in[1];
  const float* bq = (const float*)d_in[2];
  const float* wk = (const float*)d_in[3];
  const float* bk = (const float*)d_in[4];
  const float* wv = (const float*)d_in[5];
  const float* bv = (const float*)d_in[6];
  const float* wo = (const float*)d_in[7];
  const float* bo = (const float*)d_in[8];

  const size_t NQ = (size_t)M_ * E_;   // 4,194,304
  const size_t NW = (size_t)E_ * E_;   //   262,144

  unsigned short* ws16 = (unsigned short*)d_ws;
  unsigned short* Xbf  = ws16;
  unsigned short* Wqb  = Xbf + NQ;
  unsigned short* Wkb  = Wqb + NW;
  unsigned short* Wvb  = Wkb + NW;
  unsigned short* Wob  = Wvb + NW;
  unsigned short* Qb   = Wob + NW;
  unsigned short* Kb   = Qb  + NQ;
  unsigned short* VTb  = Kb  + NQ;
  unsigned short* Attn = VTb + NQ;     // total ~44 MB of ws

  cvt_bf16_kernel<<<2048, 256, 0, stream>>>(q,  Xbf, (int)NQ);
  cvt_bf16_kernel<<<512,  256, 0, stream>>>(wq, Wqb, (int)NW);
  cvt_bf16_kernel<<<512,  256, 0, stream>>>(wk, Wkb, (int)NW);
  cvt_bf16_kernel<<<512,  256, 0, stream>>>(wv, Wvb, (int)NW);
  cvt_bf16_kernel<<<512,  256, 0, stream>>>(wo, Wob, (int)NW);

  dim3 gp(M_ / 32, 2, 3);   // 256 x 2 x 3
  proj_qkv_kernel<<<gp, 128, 0, stream>>>(Xbf, Wqb, Wkb, Wvb, bq, bk, bv,
                                          Qb, Kb, VTb);

  dim3 gf(N_ / 64, H_, B_); // 32 x 8 x 4
  flash_attn_kernel<<<gf, 128, 0, stream>>>(Qb, Kb, VTb, Attn);

  dim3 go(M_ / 32, 2, 1);   // 256 x 2
  out_proj_kernel<<<go, 128, 0, stream>>>(Attn, Wob, bo, (float*)d_out);
}
